// SymmetricBilinearReduction_19748259627283
// MI455X (gfx1250) — matrix operand order repaired and checked by reading, not compiled, not run on hardware
//
#include <hip/hip_runtime.h>

// ---- problem constants ----
#define NB   8
#define K1   2048
#define K2   2048
#define EMB  512
#define RED  64
// emb_scale^2 * red_scale = (1/512) * (1/8)
#define SCORE_SCALE (1.0f / 4096.0f)

typedef __attribute__((ext_vector_type(16))) __bf16 bf16x16;
typedef __attribute__((ext_vector_type(8)))  __bf16 bf16x8;
typedef __attribute__((ext_vector_type(8)))  float  f32x8;
typedef __attribute__((ext_vector_type(4)))  float  f32x4;

static __device__ __forceinline__ bf16x16 join8(bf16x8 lo, bf16x8 hi) {
    return __builtin_shufflevector(lo, hi, 0,1,2,3,4,5,6,7,8,9,10,11,12,13,14,15);
}

// ---------------------------------------------------------------------------
// Stage 0: pack R (512x64 fp32, row-major) into bf16 B-fragment-major layout:
//   Rp[((kc*64 + n) * 32) + j] = R[(kc*32 + j)*64 + n],  kc=0..15, j=0..31
// so a WMMA B-fragment lane reads two contiguous 8-bf16 (16 B) runs, 16 k apart.
// ---------------------------------------------------------------------------
__global__ __launch_bounds__(256) void packR_kernel(const float* __restrict__ Rm,
                                                    __bf16* __restrict__ Rp) {
    const int id = blockIdx.x * 256 + threadIdx.x;   // 0 .. 32767
    const int j  = id & 31;                          // k within 32-chunk
    const int n  = (id >> 5) & 63;                   // column
    const int kc = id >> 11;                         // k chunk
    Rp[id] = (__bf16)Rm[(size_t)(kc * 32 + j) * RED + n];
}

// ---------------------------------------------------------------------------
// Stage 1: project X(16384x512) @ R -> bf16, compile-time-specialized on
// whether this is the embeddings_b half (IS_B fuses the zero-row mask).
// One block = 16 rows; 4 waves, each owning one 16-col tile.
// Lane pair (L, L+16) jointly touches every K of its row, so the all-zero
// test for embeddings_b rows is computed for free and wave 0 emits colbias.
// ---------------------------------------------------------------------------
template <bool IS_B>
__global__ __launch_bounds__(128) void project_kernel(const float* __restrict__ X,
                                                      const __bf16* __restrict__ Rp,
                                                      const float* __restrict__ bptr,
                                                      __bf16* __restrict__ Y,
                                                      float* __restrict__ colbias) {
    const int lane = threadIdx.x & 31;
    const int wv   = threadIdx.x >> 5;          // 0..3 -> col tile
    const size_t row0 = (size_t)blockIdx.x * 16;

    const int m   = lane & 15;
    const int ko  = (lane < 16) ? 0 : 8;        // A-frag K sub-offset
    const int kbo = (lane < 16) ? 0 : 16;       // unused (B frag also uses ko)
    const int n0  = wv * 16;
    const float* xrow = X + (row0 + m) * (size_t)EMB;
    const __bf16* rcol = Rp + (size_t)(n0 + m) * 32 + ko;

    int nz = 0;
    f32x8 acc = {};
    for (int kc = 0; kc < 16; ++kc) {
        const int kb = kc * 32;
        // A fragment: two contiguous 8-element K chunks
        f32x4 x0 = *(const f32x4*)(xrow + kb + ko);
        f32x4 x1 = *(const f32x4*)(xrow + kb + ko + 4);
        f32x4 x2 = *(const f32x4*)(xrow + kb + ko + 16);
        f32x4 x3 = *(const f32x4*)(xrow + kb + ko + 20);
        if (IS_B) {
            #pragma unroll
            for (int i = 0; i < 4; ++i)
                nz |= (x0[i] != 0.0f) | (x1[i] != 0.0f) | (x2[i] != 0.0f) | (x3[i] != 0.0f);
        }
        bf16x16 af;
        #pragma unroll
        for (int i = 0; i < 4; ++i) {
            af[i]      = (__bf16)x0[i];
            af[4 + i]  = (__bf16)x1[i];
            af[8 + i]  = (__bf16)x2[i];
            af[12 + i] = (__bf16)x3[i];
        }
        // B fragment: two 8-element K chunks from packed R (matches A)
        const __bf16* bp = rcol + (size_t)kc * (64 * 32);
        bf16x16 bfrag = join8(*(const bf16x8*)(bp), *(const bf16x8*)(bp + 16));

        acc = __builtin_amdgcn_wmma_f32_16x16x32_bf16(false, af, false, bfrag,
                                                      (short)0, acc, false, false);
    }
    #pragma unroll
    for (int v = 0; v < 8; ++v) {
        const int row = v + ((lane >= 16) ? 8 : 0);
        Y[(row0 + row) * RED + n0 + m] = (__bf16)acc[v];
    }
    // fused colbias: bias - 1e9 * (row all zero); lanes L and L+16 cover all K
    if (IS_B) {
        if (wv == 0) {
            nz |= __shfl_xor(nz, 16, 32);
            if (lane < 16)
                colbias[row0 + lane] = bptr[0] - (nz ? 0.0f : 1e9f);
        }
    }
}

// ---------------------------------------------------------------------------
// Stage 2: scores = ra @ rb^T * SCALE + colbias, fused row softmax.
// Block = (batch, 16-row stripe); 8 waves x 16 col-tiles = full 2048 cols.
// Entire 16x2048 score stripe stays in registers (16 x v8f per lane).
// Output stored non-temporally (write-once 134 MB stream).
// ---------------------------------------------------------------------------
__global__ __launch_bounds__(256) void scores_kernel(const __bf16* __restrict__ ra,
                                                     const __bf16* __restrict__ rb,
                                                     const float* __restrict__ colbias,
                                                     float* __restrict__ out) {
    const int lane  = threadIdx.x & 31;
    const int wave  = threadIdx.x >> 5;          // 0..7
    const int batch = blockIdx.y;                // 0..7
    const int m0    = blockIdx.x * 16;           // row stripe base

    const __bf16* raB = ra + ((size_t)batch * K1 + m0) * RED;
    const __bf16* rbB = rb + (size_t)batch * K2 * RED;
    const float*  cbB = colbias + (size_t)batch * K2;

    const int m  = lane & 15;
    const int hi = (lane >= 16) ? 1 : 0;
    const int ko = hi ? 8 : 0;

    // A fragments for the stripe's 16 rows, K=0..31 and K=32..63
    const __bf16* arow = raB + m * RED;
    bf16x16 a0 = join8(*(const bf16x8*)(arow + ko),      *(const bf16x8*)(arow + ko + 16));
    bf16x16 a1 = join8(*(const bf16x8*)(arow + 32 + ko), *(const bf16x8*)(arow + 32 + ko + 16));

    f32x8 acc[16];
    #pragma unroll
    for (int t = 0; t < 16; ++t) {
        const int n0 = (wave + t * 8) * 16;
        const __bf16* brow = rbB + (size_t)(n0 + m) * RED + ko;
        bf16x16 b0 = join8(*(const bf16x8*)(brow),      *(const bf16x8*)(brow + 16));
        bf16x16 b1 = join8(*(const bf16x8*)(brow + 32), *(const bf16x8*)(brow + 48));

        f32x8 c = {};
        c = __builtin_amdgcn_wmma_f32_16x16x32_bf16(false, a0, false, b0, (short)0, c, false, false);
        c = __builtin_amdgcn_wmma_f32_16x16x32_bf16(false, a1, false, b1, (short)0, c, false, false);

        const float cb = cbB[n0 + m];
        #pragma unroll
        for (int v = 0; v < 8; ++v) acc[t][v] = c[v] * SCORE_SCALE + cb;
    }

    __shared__ float s_red[8][16];
    __shared__ float s_row[16];

    // ---- pass 1: row max ----
    float mx[8];
    #pragma unroll
    for (int v = 0; v < 8; ++v) {
        float x = acc[0][v];
        #pragma unroll
        for (int t = 1; t < 16; ++t) x = fmaxf(x, acc[t][v]);
        #pragma unroll
        for (int msk = 1; msk <= 8; msk <<= 1) x = fmaxf(x, __shfl_xor(x, msk, 32));
        mx[v] = x;
    }
    if ((lane & 15) == 0) {
        #pragma unroll
        for (int v = 0; v < 8; ++v) s_red[wave][v + hi * 8] = mx[v];
    }
    __syncthreads();
    if (threadIdx.x < 16) {
        float x = s_red[0][threadIdx.x];
        #pragma unroll
        for (int w = 1; w < 8; ++w) x = fmaxf(x, s_red[w][threadIdx.x]);
        s_row[threadIdx.x] = x;
    }
    __syncthreads();
    float rowmax[8];
    #pragma unroll
    for (int v = 0; v < 8; ++v) rowmax[v] = s_row[v + hi * 8];

    // ---- pass 2: exp + row sum ----
    float sm[8];
    #pragma unroll
    for (int v = 0; v < 8; ++v) sm[v] = 0.0f;
    #pragma unroll
    for (int t = 0; t < 16; ++t) {
        #pragma unroll
        for (int v = 0; v < 8; ++v) {
            float e = __expf(acc[t][v] - rowmax[v]);
            acc[t][v] = e;
            sm[v] += e;
        }
    }
    #pragma unroll
    for (int v = 0; v < 8; ++v) {
        float x = sm[v];
        #pragma unroll
        for (int msk = 1; msk <= 8; msk <<= 1) x += __shfl_xor(x, msk, 32);
        sm[v] = x;
    }
    if ((lane & 15) == 0) {
        #pragma unroll
        for (int v = 0; v < 8; ++v) s_red[wave][v + hi * 8] = sm[v];
    }
    __syncthreads();
    if (threadIdx.x < 16) {
        float x = 0.0f;
        #pragma unroll
        for (int w = 0; w < 8; ++w) x += s_red[w][threadIdx.x];
        s_row[threadIdx.x] = 1.0f / x;
    }
    __syncthreads();
    float rinv[8];
    #pragma unroll
    for (int v = 0; v < 8; ++v) rinv[v] = s_row[v + hi * 8];

    // ---- pass 3: normalized write, non-temporal (only HBM write of scores) ----
    float* outB = out + ((size_t)batch * K1 + m0) * (size_t)K2;
    #pragma unroll
    for (int t = 0; t < 16; ++t) {
        const int n0 = (wave + t * 8) * 16;
        #pragma unroll
        for (int v = 0; v < 8; ++v) {
            const int row = v + hi * 8;
            __builtin_nontemporal_store(acc[t][v] * rinv[v],
                                        outB + (size_t)row * K2 + n0 + m);
        }
    }
}

// ---------------------------------------------------------------------------
extern "C" void kernel_launch(void* const* d_in, const int* in_sizes, int n_in,
                              void* d_out, int out_size, void* d_ws, size_t ws_size,
                              hipStream_t stream) {
    const float* embA = (const float*)d_in[0];
    const float* embB = (const float*)d_in[1];
    const float* Rm   = (const float*)d_in[2];
    const float* bptr = (const float*)d_in[3];
    float* out = (float*)d_out;

    char* ws = (char*)d_ws;
    const size_t raBytes = (size_t)NB * K1 * RED * sizeof(__bf16);  // 2 MB
    __bf16* ra      = (__bf16*)ws;
    __bf16* rb      = (__bf16*)(ws + raBytes);
    float*  colbias = (float*)(ws + 2 * raBytes);                   // 64 KB
    __bf16* Rp      = (__bf16*)(ws + 2 * raBytes + (size_t)NB * K2 * sizeof(float));

    packR_kernel<<<(EMB * RED) / 256, 256, 0, stream>>>(Rm, Rp);
    project_kernel<false><<<(NB * K1) / 16, 128, 0, stream>>>(embA, Rp, bptr,
                                                              ra, colbias);
    project_kernel<true><<<(NB * K2) / 16, 128, 0, stream>>>(embB, Rp, bptr,
                                                             rb, colbias);
    scores_kernel<<<dim3(K1 / 16, NB), 256, 0, stream>>>(ra, rb, colbias, out);
}
